// GRUwLinearModel_34548716929350
// MI455X (gfx1250) — compile-verified, matrix operand order checked by reading, not run on hardware
//
#include <hip/hip_runtime.h>

// GRU + dynamic linear readout, persistent-kernel formulation for MI455X (gfx1250).
// B=128, T=2048, IN=32, H=256, L=16.
//
// 8 workgroups x 512 threads (16 waves, wave32). Workgroup g owns batch rows
// [16g, 16g+16). Wave j owns hidden-column slice [16j, 16j+16).
// Register/LDS split (to stay under ~256 VGPRs, avoiding scratch spills):
//   - Whh fragments for r,z gates: resident in VGPRs (128 regs)
//   - Whh fragments for n gate + all Wih fragments: staged in LDS in the
//     exact per-lane fragment layout, reloaded via 2x ds_load_b128 per frag
//   - h state: LDS bf16 ping-pong buffers; one barrier per timestep

#define B_  128
#define T_  2048
#define IN_ 32
#define H_  256
#define L_  16

#define NWAVE   16
#define NLDSF   11           // per-wave frags in LDS: Bn[0..7], Brx, Bzx, Bnx
#define HBUF_E  (2 * 16 * H_)                  // ushort elements
#define BSTORE_E (NWAVE * NLDSF * 32 * 16)     // ushort elements
#define SMEM_BYTES ((HBUF_E + BSTORE_E) * 2)   // 16KB + 176KB = 192KB

typedef __attribute__((ext_vector_type(16))) __bf16 v16bf;
typedef __attribute__((ext_vector_type(8)))  float  v8f;

union Frag {
    v16bf v;
    unsigned short u[16];
    uint4 q[2];
};

__device__ __forceinline__ unsigned short f2bf(float f) {
    unsigned x = __float_as_uint(f);
    unsigned r = (x + 0x7FFFu + ((x >> 16) & 1u)) >> 16;  // round-to-nearest-even
    return (unsigned short)r;
}
__device__ __forceinline__ float bf2f(unsigned short h) {
    return __uint_as_float(((unsigned)h) << 16);
}
__device__ __forceinline__ float sigmoid_(float x) {
    return 1.0f / (1.0f + __expf(-x));
}
__device__ __forceinline__ float tanh_(float x) {
    return 1.0f - 2.0f / (__expf(2.0f * x) + 1.0f);  // well-behaved at +-inf
}

// B-fragment (32x16 KxN, bf16) from row-major W[outRow][k] with stride S.
// Lane L -> n = L&15, half = L>>4; element e -> K = (e>=8?16:0)+8*half+(e&7).
__device__ __forceinline__ v16bf load_bfrag(const float* __restrict__ W, int S,
                                            int colbase, int koff, int n, int half) {
    Frag f;
    const float* p = W + (size_t)(colbase + n) * S + koff + 8 * half;
#pragma unroll
    for (int e = 0; e < 8; ++e) f.u[e] = f2bf(p[e]);
#pragma unroll
    for (int e = 0; e < 8; ++e) f.u[8 + e] = f2bf(p[16 + e]);
    return f.v;
}

// Stage/reload fragments in LDS, packed per (wave, frag, lane) as 32B chunks.
__device__ __forceinline__ void lds_store_frag(unsigned short* bstore, int wave,
                                               int f, int lane, const Frag& fr) {
    uint4* dst = reinterpret_cast<uint4*>(bstore + ((wave * NLDSF + f) * 32 + lane) * 16);
    dst[0] = fr.q[0];
    dst[1] = fr.q[1];
}
__device__ __forceinline__ v16bf lds_load_frag(const unsigned short* bstore, int wave,
                                               int f, int lane) {
    Frag fr;
    const uint4* src =
        reinterpret_cast<const uint4*>(bstore + ((wave * NLDSF + f) * 32 + lane) * 16);
    fr.q[0] = src[0];
    fr.q[1] = src[1];
    return fr.v;
}

// A-fragment (16x32 MxK, bf16) of the current x_t tile from global f32 (float4 loads).
__device__ __forceinline__ v16bf load_x_frag(const float* __restrict__ xg,
                                             int rowb, int t, int m, int half) {
    Frag f;
    const float* p = xg + ((size_t)(rowb + m) * T_ + t) * IN_ + 8 * half;
    const float4* p4 = reinterpret_cast<const float4*>(p);
    float4 a = p4[0], b = p4[1];   // K = 8*half + 0..7
    float4 c = p4[4], d = p4[5];   // K = 16 + 8*half + 0..7
    f.u[0] = f2bf(a.x); f.u[1] = f2bf(a.y); f.u[2]  = f2bf(a.z); f.u[3]  = f2bf(a.w);
    f.u[4] = f2bf(b.x); f.u[5] = f2bf(b.y); f.u[6]  = f2bf(b.z); f.u[7]  = f2bf(b.w);
    f.u[8] = f2bf(c.x); f.u[9] = f2bf(c.y); f.u[10] = f2bf(c.z); f.u[11] = f2bf(c.w);
    f.u[12] = f2bf(d.x); f.u[13] = f2bf(d.y); f.u[14] = f2bf(d.z); f.u[15] = f2bf(d.w);
    return f.v;
}

// A-fragment of h from LDS (bf16 row-major [16][256]); two ds_load_b128 per lane.
__device__ __forceinline__ v16bf load_h_frag(const unsigned short* hb, int kb,
                                             int m, int half) {
    Frag f;
    const unsigned short* p = hb + m * H_ + kb * 32 + 8 * half;
    f.q[0] = *reinterpret_cast<const uint4*>(p);
    f.q[1] = *reinterpret_cast<const uint4*>(p + 16);
    return f.v;
}

__device__ __forceinline__ v8f wmma_bf16(v16bf a, v16bf b, v8f c) {
    return __builtin_amdgcn_wmma_f32_16x16x32_bf16(false, a, false, b,
                                                   (short)0, c, false, false);
}

__global__ __launch_bounds__(512, 1)
void gru_linear_persistent(const float* __restrict__ input_gru,
                           const float* __restrict__ input_linear,
                           const float* __restrict__ init_hidden,
                           const float* __restrict__ Wih,
                           const float* __restrict__ Whh,
                           const float* __restrict__ bias,
                           const float* __restrict__ bias_n,
                           float* __restrict__ out) {
    extern __shared__ __align__(16) unsigned short smem[];
    unsigned short* hbuf   = smem;             // [2][16*H_] bf16 ping-pong
    unsigned short* bstore = smem + HBUF_E;    // [NWAVE][NLDSF][32][16] fragments

    const int tid  = threadIdx.x;
    const int wave = tid >> 5;        // 0..15: hidden-column slice owner
    const int lane = tid & 31;
    const int mn   = lane & 15;       // A: batch row m; B/D: column n
    const int half = lane >> 4;
    const int rowb = blockIdx.x * 16; // batch tile base
    const int col  = wave * 16 + mn;  // hidden column owned by this lane

    // ---- VGPR-resident B-fragments: r,z gates of Whh (K=256 -> 8 frags each) ----
    v16bf Br[8], Bz[8];
#pragma unroll
    for (int kb = 0; kb < 8; ++kb) {
        Br[kb] = load_bfrag(Whh, H_, 0  + wave * 16, kb * 32, mn, half);
        Bz[kb] = load_bfrag(Whh, H_, H_ + wave * 16, kb * 32, mn, half);
    }
    // ---- LDS-staged fragments: n gate of Whh + all three Wih fragments ----
    for (int kb = 0; kb < 8; ++kb) {
        Frag f; f.v = load_bfrag(Whh, H_, 2 * H_ + wave * 16, kb * 32, mn, half);
        lds_store_frag(bstore, wave, kb, lane, f);
    }
    {
        Frag f;
        f.v = load_bfrag(Wih, IN_, 0      + wave * 16, 0, mn, half);
        lds_store_frag(bstore, wave, 8, lane, f);
        f.v = load_bfrag(Wih, IN_, H_     + wave * 16, 0, mn, half);
        lds_store_frag(bstore, wave, 9, lane, f);
        f.v = load_bfrag(Wih, IN_, 2 * H_ + wave * 16, 0, mn, half);
        lds_store_frag(bstore, wave, 10, lane, f);
    }

    // Per-lane bias scalars (bias depends only on the N column of the D tile).
    const float b_r  = bias[0      + col];
    const float b_z  = bias[H_     + col];
    const float b_in = bias[2 * H_ + col];
    const float bn   = bias_n[col];

    // ---- h0 -> LDS buffer 0 (cooperative, bf16) ----
    for (int i = tid; i < 16 * H_; i += 512) {
        int m = i >> 8, k = i & (H_ - 1);
        hbuf[i] = f2bf(init_hidden[(size_t)(rowb + m) * H_ + k]);
    }
    __syncthreads();

    for (int t = 0; t < T_; ++t) {
        const int cur = t & 1, nxt = cur ^ 1;

        // Prefetch next timestep's inputs (global_prefetch_b8).
        if (t + 1 < T_) {
            __builtin_prefetch(input_gru + ((size_t)(rowb + mn) * T_ + (t + 1)) * IN_, 0, 1);
            if (wave == 0)
                __builtin_prefetch(input_linear + ((size_t)(rowb + mn) * T_ + (t + 1)) * L_, 0, 1);
        }

        // Accumulators; r/z/i_n start from bias (input-side bias folded in),
        // h_n kept separate because n = tanh(i_n + r*(h_n + bias_n)).
        v8f r_acc{}, z_acc{}, hn_acc{}, in_acc{};
#pragma unroll
        for (int v = 0; v < 8; ++v) {
            r_acc[v] = b_r; z_acc[v] = b_z; in_acc[v] = b_in; hn_acc[v] = 0.0f;
        }

        // Recurrent GEMM: [16 x 256] x [256 x 16] per gate, K-blocked by 32.
#pragma unroll
        for (int kb = 0; kb < 8; ++kb) {
            v16bf a  = load_h_frag(&hbuf[cur * 16 * H_], kb, mn, half);
            v16bf bn_f = lds_load_frag(bstore, wave, kb, lane);
            r_acc  = wmma_bf16(a, Br[kb], r_acc);
            z_acc  = wmma_bf16(a, Bz[kb], z_acc);
            hn_acc = wmma_bf16(a, bn_f,   hn_acc);
        }
        // Input-side contribution (K = IN = 32, one WMMA per gate).
        {
            v16bf ax  = load_x_frag(input_gru, rowb, t, mn, half);
            v16bf brx = lds_load_frag(bstore, wave, 8, lane);
            v16bf bzx = lds_load_frag(bstore, wave, 9, lane);
            v16bf bnx = lds_load_frag(bstore, wave, 10, lane);
            r_acc  = wmma_bf16(ax, brx, r_acc);
            z_acc  = wmma_bf16(ax, bzx, z_acc);
            in_acc = wmma_bf16(ax, bnx, in_acc);
        }

        // Elementwise gates + state update. D[m][n]: m = v + 8*half, n = mn.
        float hnew_v[8];
#pragma unroll
        for (int v = 0; v < 8; ++v) {
            const int m = v + 8 * half;
            float r  = sigmoid_(r_acc[v]);
            float z  = sigmoid_(z_acc[v]);
            float ng = tanh_(in_acc[v] + r * (hn_acc[v] + bn));
            float ho = bf2f(hbuf[cur * 16 * H_ + m * H_ + col]);
            float hv = ng + z * (ho - ng);
            hnew_v[v] = hv;
            hbuf[nxt * 16 * H_ + m * H_ + col] = f2bf(hv);
        }

        // Readout: wave 0 owns h[:, 0:16] == dynamic linear params.
        if (wave == 0) {
#pragma unroll
            for (int v = 0; v < 8; ++v) {
                const int m = v + 8 * half;
                float p = hnew_v[v] *
                          input_linear[((size_t)(rowb + m) * T_ + t) * L_ + mn];
                p += __shfl_xor(p, 1, 32);
                p += __shfl_xor(p, 2, 32);
                p += __shfl_xor(p, 4, 32);
                p += __shfl_xor(p, 8, 32);
                if (mn == 0) out[(size_t)(rowb + m) * T_ + t] = p;
            }
        }
        __syncthreads();  // h(t+1) buffer complete before next step reads it
    }
}

extern "C" void kernel_launch(void* const* d_in, const int* in_sizes, int n_in,
                              void* d_out, int out_size, void* d_ws, size_t ws_size,
                              hipStream_t stream) {
    (void)in_sizes; (void)n_in; (void)out_size; (void)d_ws; (void)ws_size;
    const float* input_gru    = (const float*)d_in[0];
    const float* input_linear = (const float*)d_in[1];
    const float* init_hidden  = (const float*)d_in[2];
    const float* Wih          = (const float*)d_in[3];
    const float* Whh          = (const float*)d_in[4];
    const float* bias         = (const float*)d_in[5];
    const float* bias_n       = (const float*)d_in[6];
    float* out = (float*)d_out;

    gru_linear_persistent<<<B_ / 16, 512, SMEM_BYTES, stream>>>(
        input_gru, input_linear, init_hidden, Wih, Whh, bias, bias_n, out);
}
